// GaussianPolicy_14620068675620
// MI455X (gfx1250) — compile-verified, matrix operand order checked by reading, not run on hardware
//
#include <hip/hip_runtime.h>

typedef __attribute__((ext_vector_type(16))) _Float16 v16h;
typedef __attribute__((ext_vector_type(8)))  _Float16 v8h;
typedef __attribute__((ext_vector_type(4)))  _Float16 v4h;
typedef __attribute__((ext_vector_type(8)))  float    v8f;
typedef __attribute__((ext_vector_type(4)))  float    v4f;

#define THRESH_ 0.8f
#define DECAY_  0.2f

constexpr int B_    = 2048;
constexpr int NIN_  = 512;
constexpr int HID_  = 1024;
constexpr int NACT_ = 32;
constexpr int XC_   = 4 * HID_;   // 4096

// ---------------- workspace layout (in _Float16 units) ----------------
constexpr size_t F16_FRAMES = 0;                                         // [5][2048][512]
constexpr size_t F16_WSTACK = F16_FRAMES + (size_t)5 * B_ * NIN_;        // [3][1024][512]
constexpr size_t F16_W11    = F16_WSTACK + (size_t)3 * HID_ * NIN_;      // [1024][4096]
constexpr size_t F16_W21    = F16_W11 + (size_t)HID_ * XC_;
constexpr size_t F16_W12    = F16_W21 + (size_t)HID_ * XC_;              // [1024][1024]
constexpr size_t F16_W22    = F16_W12 + (size_t)HID_ * HID_;
constexpr size_t F16_WM     = F16_W22 + (size_t)HID_ * HID_;             // [32][1024]
constexpr size_t F16_WSW    = F16_WM + (size_t)NACT_ * HID_;
constexpr size_t F16_X      = F16_WSW + (size_t)NACT_ * HID_;            // [2048][4096]
constexpr size_t F16_T1     = F16_X + (size_t)B_ * XC_;                  // [2048][1024]
constexpr size_t F16_T2     = F16_T1 + (size_t)B_ * HID_;
constexpr size_t F16_T3     = F16_T2 + (size_t)B_ * HID_;
constexpr size_t F16_T4     = F16_T3 + (size_t)B_ * HID_;

// ---------------- WMMA operand loader ----------------
// Operand for v_wmma_f32_16x16x32_f16 from a row-major [16 x K] tile.
// Lanes 0-15 hold row M=lane: halves 0..7 = K 0..7, halves 8..15 = K 16..23;
// lanes 16-31 hold the same rows shifted by K+8 / K+24.
__device__ __forceinline__ v16h load_op16(const _Float16* __restrict__ base,
                                          int row0, int ld, int k0, int lane) {
    const int l  = lane & 15;
    const int hi = (lane >> 4) & 1;
    const _Float16* p = base + (size_t)(row0 + l) * ld + (k0 + hi * 8);
    v8h lo = *(const v8h*)(p);
    v8h hh = *(const v8h*)(p + 16);
    v16h r;
#pragma unroll
    for (int i = 0; i < 8; ++i) { r[i] = lo[i]; r[i + 8] = hh[i]; }
    return r;
}

__device__ __forceinline__ v8f wmma_f16(v16h a, v16h b, v8f c) {
    return __builtin_amdgcn_wmma_f32_16x16x32_f16(false, a, false, b,
                                                  (short)0, c, false, false);
}

// ---------------- conversion / packing kernels ----------------
__global__ void cvt_f32_f16_k(const float* __restrict__ in,
                              _Float16* __restrict__ out, int n4) {
    int i = blockIdx.x * blockDim.x + threadIdx.x;
    if (i >= n4) return;
    v4f v = ((const v4f*)in)[i];
    v4h h; h[0] = (_Float16)v[0]; h[1] = (_Float16)v[1];
    h[2] = (_Float16)v[2]; h[3] = (_Float16)v[3];
    ((v4h*)out)[i] = h;
}

// state [B,5,NIN] f32  ->  frames [5][B][NIN] f16
__global__ void pack_frames_k(const float* __restrict__ state,
                              _Float16* __restrict__ frames) {
    int i4 = blockIdx.x * blockDim.x + threadIdx.x;
    const int total4 = 5 * B_ * NIN_ / 4;
    if (i4 >= total4) return;
    int i   = i4 * 4;
    int f   = i / (B_ * NIN_);
    int rem = i - f * (B_ * NIN_);
    int b   = rem / NIN_;
    int k   = rem - b * NIN_;
    v4f v = *(const v4f*)(state + (size_t)b * 5 * NIN_ + (size_t)f * NIN_ + k);
    v4h h; h[0] = (_Float16)v[0]; h[1] = (_Float16)v[1];
    h[2] = (_Float16)v[2]; h[3] = (_Float16)v[3];
    *(v4h*)(frames + i) = h;
}

// ---------------- fused input-GEMM + spiking scan ----------------
// Block = 256 threads = 8 waves arranged 2(m) x 4(n). Wave tile: 32 batch x
// 16 hid x 3 channels -> 6 WMMA accumulators; per k-step of 32: 10 b128 loads
// feed 6 WMMAs. Membrane state + spike sums (128 VGPRs) stay in registers
// across all 5 frames (spike is recomputable as mem > THRESH).
__global__ __launch_bounds__(256)
void scan_k(const _Float16* __restrict__ frames,
            const _Float16* __restrict__ wstack,
            const float* __restrict__ b1, const float* __restrict__ b2,
            const float* __restrict__ b3,
            const float* __restrict__ wl, const float* __restrict__ bl,
            _Float16* __restrict__ x) {
    const int lane = threadIdx.x & 31;
    const int wave = threadIdx.x >> 5;
    const int nl   = lane & 15;
    const int m0   = blockIdx.x * 64 + (wave >> 2) * 32;   // batch (2 m-tiles)
    const int h0   = blockIdx.y * 64 + (wave & 3) * 16;    // hid

    const float wl0 = wl[0], wl1 = wl[1], wl2 = wl[2], blv = bl[0];
    const float bb[3] = { b1[h0 + nl], b2[h0 + nl], b3[h0 + nl] };

    float mem[16][4], ssum[16][4];
#pragma unroll
    for (int s = 0; s < 16; ++s)
#pragma unroll
        for (int k = 0; k < 4; ++k) { mem[s][k] = 0.f; ssum[s][k] = 0.f; }

    const _Float16* W[3] = { wstack,
                             wstack + (size_t)HID_ * NIN_,
                             wstack + (size_t)2 * HID_ * NIN_ };

    for (int f = 0; f < 5; ++f) {
        v8f acc[3][2];
#pragma unroll
        for (int c = 0; c < 3; ++c)
#pragma unroll
            for (int mt = 0; mt < 2; ++mt) acc[c][mt] = (v8f){};

        const _Float16* Af = frames + (size_t)f * B_ * NIN_;
#pragma unroll 2
        for (int kk = 0; kk < NIN_; kk += 32) {
            v16h a0 = load_op16(Af, m0,      NIN_, kk, lane);
            v16h a1 = load_op16(Af, m0 + 16, NIN_, kk, lane);
#pragma unroll
            for (int c = 0; c < 3; ++c) {
                v16h b = load_op16(W[c], h0, NIN_, kk, lane);
                acc[c][0] = wmma_f16(a0, b, acc[c][0]);
                acc[c][1] = wmma_f16(a1, b, acc[c][1]);
            }
        }
        // 3 repeated timesteps of this frame, fully in registers.
#pragma unroll
        for (int rep = 0; rep < 3; ++rep) {
#pragma unroll
            for (int mt = 0; mt < 2; ++mt) {
#pragma unroll
                for (int r = 0; r < 8; ++r) {
                    const int s = mt * 8 + r;
                    float inner = mem[s][0] * wl0 + mem[s][1] * wl1 +
                                  mem[s][2] * wl2 + blv;
                    float inp[4];
                    inp[0] = acc[0][mt][r] + bb[0];
                    inp[1] = acc[1][mt][r] + bb[1];
                    inp[2] = acc[2][mt][r] + bb[2];
                    inp[3] = inner;
#pragma unroll
                    for (int k = 0; k < 4; ++k) {
                        float sp = (mem[s][k] > THRESH_) ? 1.f : 0.f;  // carried spike
                        float m2 = mem[s][k] * DECAY_ * (1.f - sp) + inp[k];
                        mem[s][k] = m2;
                        ssum[s][k] += (m2 > THRESH_) ? 1.f : 0.f;
                    }
                }
            }
        }
    }
    // x[b, h*4 + k] = mean over 15 steps, stored f16, one 8B store per slot.
    const float inv15 = 1.f / 15.f;
#pragma unroll
    for (int mt = 0; mt < 2; ++mt) {
#pragma unroll
        for (int r = 0; r < 8; ++r) {
            const int s  = mt * 8 + r;
            int bi = m0 + mt * 16 + ((lane < 16) ? r : r + 8);
            int h  = h0 + nl;
            v4h o;
#pragma unroll
            for (int k = 0; k < 4; ++k) o[k] = (_Float16)(ssum[s][k] * inv15);
            *(v4h*)(x + (size_t)bi * XC_ + (size_t)h * 4) = o;
        }
    }
}

// ---------------- generic WMMA GEMM: out = act(A @ W^T + bias) ----------------
// MODE 0: relu -> f16 out   MODE 1: f32 out   MODE 2: clip[-20,2] -> f32 out
// Block = 8 waves arranged WM x WN; each wave computes (MT*16) x (NT*16).
template <int MODE, int WM, int WN, int MT, int NT>
__global__ __launch_bounds__(256)
void gemm_k(const _Float16* __restrict__ A, const _Float16* __restrict__ W,
            const float* __restrict__ bias, void* __restrict__ out,
            int N, int K) {
    const int lane = threadIdx.x & 31;
    const int wave = threadIdx.x >> 5;
    const int nl   = lane & 15;
    const int wm   = wave / WN, wn = wave % WN;
    const int m0   = (blockIdx.x * WM + wm) * (MT * 16);
    const int n0   = (blockIdx.y * WN + wn) * (NT * 16);

    v8f acc[MT][NT];
#pragma unroll
    for (int mt = 0; mt < MT; ++mt)
#pragma unroll
        for (int t = 0; t < NT; ++t) acc[mt][t] = (v8f){};

#pragma unroll 2
    for (int kk = 0; kk < K; kk += 32) {
        v16h a[MT];
#pragma unroll
        for (int mt = 0; mt < MT; ++mt)
            a[mt] = load_op16(A, m0 + mt * 16, K, kk, lane);
#pragma unroll
        for (int t = 0; t < NT; ++t) {
            v16h b = load_op16(W, n0 + t * 16, K, kk, lane);
#pragma unroll
            for (int mt = 0; mt < MT; ++mt)
                acc[mt][t] = wmma_f16(a[mt], b, acc[mt][t]);
        }
    }

#pragma unroll
    for (int t = 0; t < NT; ++t) {
        int   col = n0 + t * 16 + nl;
        float bv  = bias[col];
#pragma unroll
        for (int mt = 0; mt < MT; ++mt) {
#pragma unroll
            for (int r = 0; r < 8; ++r) {
                int   row = m0 + mt * 16 + ((lane < 16) ? r : r + 8);
                float v   = acc[mt][t][r] + bv;
                if (MODE == 0) {
                    v = v > 0.f ? v : 0.f;
                    ((_Float16*)out)[(size_t)row * N + col] = (_Float16)v;
                } else if (MODE == 1) {
                    ((float*)out)[(size_t)row * N + col] = v;
                } else {
                    v = fminf(fmaxf(v, -20.f), 2.f);
                    ((float*)out)[(size_t)row * N + col] = v;
                }
            }
        }
    }
}

// ---------------- host launcher ----------------
extern "C" void kernel_launch(void* const* d_in, const int* in_sizes, int n_in,
                              void* d_out, int out_size, void* d_ws, size_t ws_size,
                              hipStream_t stream) {
    const float* state = (const float*)d_in[0];
    const float* w1  = (const float*)d_in[1];  const float* b1  = (const float*)d_in[2];
    const float* w2  = (const float*)d_in[3];  const float* b2  = (const float*)d_in[4];
    const float* w3  = (const float*)d_in[5];  const float* b3  = (const float*)d_in[6];
    const float* wl  = (const float*)d_in[7];  const float* bl  = (const float*)d_in[8];
    const float* w11 = (const float*)d_in[9];  const float* b11 = (const float*)d_in[10];
    const float* w12 = (const float*)d_in[11]; const float* b12 = (const float*)d_in[12];
    const float* w21 = (const float*)d_in[13]; const float* b21 = (const float*)d_in[14];
    const float* w22 = (const float*)d_in[15]; const float* b22 = (const float*)d_in[16];
    const float* wm  = (const float*)d_in[17]; const float* bm  = (const float*)d_in[18];
    const float* wsw = (const float*)d_in[19]; const float* bs  = (const float*)d_in[20];

    _Float16* wsp = (_Float16*)d_ws;
    _Float16* frames = wsp + F16_FRAMES;
    _Float16* wstack = wsp + F16_WSTACK;
    _Float16* w11f   = wsp + F16_W11;
    _Float16* w21f   = wsp + F16_W21;
    _Float16* w12f   = wsp + F16_W12;
    _Float16* w22f   = wsp + F16_W22;
    _Float16* wmf    = wsp + F16_WM;
    _Float16* wsf    = wsp + F16_WSW;
    _Float16* x      = wsp + F16_X;
    _Float16* t1     = wsp + F16_T1;
    _Float16* t2     = wsp + F16_T2;
    _Float16* t3     = wsp + F16_T3;
    _Float16* t4     = wsp + F16_T4;

    const int TB = 256;
    auto cvt = [&](const float* src, _Float16* dst, size_t n) {
        int n4 = (int)(n / 4);
        cvt_f32_f16_k<<<(n4 + TB - 1) / TB, TB, 0, stream>>>(src, dst, n4);
    };

    // pack / convert
    {
        int n4 = 5 * B_ * NIN_ / 4;
        pack_frames_k<<<(n4 + TB - 1) / TB, TB, 0, stream>>>(state, frames);
    }
    cvt(w1,  wstack,                         (size_t)HID_ * NIN_);
    cvt(w2,  wstack + (size_t)HID_ * NIN_,   (size_t)HID_ * NIN_);
    cvt(w3,  wstack + (size_t)2*HID_*NIN_,   (size_t)HID_ * NIN_);
    cvt(w11, w11f, (size_t)HID_ * XC_);
    cvt(w21, w21f, (size_t)HID_ * XC_);
    cvt(w12, w12f, (size_t)HID_ * HID_);
    cvt(w22, w22f, (size_t)HID_ * HID_);
    cvt(wm,  wmf,  (size_t)NACT_ * HID_);
    cvt(wsw, wsf,  (size_t)NACT_ * HID_);

    // fused input GEMM + spiking scan -> x [2048 x 4096] f16
    scan_k<<<dim3(B_ / 64, HID_ / 64), 256, 0, stream>>>(
        frames, wstack, b1, b2, b3, wl, bl, x);

    // hidden layers: relu(A @ W^T + b) -> f16 ; block tile 128x128
    gemm_k<0, 4, 2, 2, 4><<<dim3(B_ / 128, HID_ / 128), 256, 0, stream>>>(
        x,  w11f, b11, (void*)t1, HID_, XC_);
    gemm_k<0, 4, 2, 2, 4><<<dim3(B_ / 128, HID_ / 128), 256, 0, stream>>>(
        t1, w12f, b12, (void*)t2, HID_, HID_);
    gemm_k<0, 4, 2, 2, 4><<<dim3(B_ / 128, HID_ / 128), 256, 0, stream>>>(
        x,  w21f, b21, (void*)t3, HID_, XC_);
    gemm_k<0, 4, 2, 2, 4><<<dim3(B_ / 128, HID_ / 128), 256, 0, stream>>>(
        t3, w22f, b22, (void*)t4, HID_, HID_);

    // heads: mean (f32) then clipped log_std (f32), concatenated in d_out
    float* out_mean = (float*)d_out;
    float* out_ls   = out_mean + (size_t)B_ * NACT_;
    gemm_k<1, 8, 1, 1, 2><<<dim3(B_ / 128, 1), 256, 0, stream>>>(
        t2, wmf, bm, (void*)out_mean, NACT_, HID_);
    gemm_k<2, 8, 1, 1, 2><<<dim3(B_ / 128, 1), 256, 0, stream>>>(
        t4, wsf, bs, (void*)out_ls, NACT_, HID_);
}
